// DMNC_78993038508758
// MI455X (gfx1250) — matrix operand order repaired and checked by reading, not compiled, not run on hardware
//
#include <hip/hip_runtime.h>
#include <hip/hip_bf16.h>
#include <math.h>

#define EMBD 64
#define MCELL 16
#define DELTAV 1e-6f
#define CLIPV 20.0f
#define L_INV 512
#define L_OUTV 64
#define TOKEN_STARTV 136
#define ENC_THREADS 128
#define ENC_WAVES 4

typedef __attribute__((ext_vector_type(16))) _Float16 v16h;
typedef __attribute__((ext_vector_type(8)))  float    v8f;

__device__ __forceinline__ float sigf(float x){ return 1.0f/(1.0f+__expf(-x)); }
__device__ __forceinline__ float sppf(float x){ float ax=fabsf(x); return fmaxf(x,0.0f)+log1pf(__expf(-ax)); }
__device__ __forceinline__ float clipf(float x){ return fminf(fmaxf(x,-CLIPV),CLIPV); }

// ---------------------------------------------------------------------------
// GEMV y = W(f16,row-major,ld) @ x(f16) + bias, via V_WMMA_F32_16X16X32_F16.
// M-tiles strided across the block's waves (wv must be an SGPR value so the
// tile loop is scalar-branched and EXEC stays all-ones for WMMA).
// A-frag = 16x32 weight tile (ISA 7.12.2 16-bit A layout), B-frag = x
// replicated into every N column; column N=0 of the f32 D tile gives y.
// Extraction: lanes 0 and 16 (N=0 halves) store 8 rows each — one predicate.
// ---------------------------------------------------------------------------
__device__ __forceinline__ void wmma_gemv(const _Float16* __restrict__ W, int ld,
                                          const _Float16* __restrict__ x,
                                          const float* __restrict__ bias,
                                          float* __restrict__ y,
                                          int mtiles, int ktiles,
                                          int lane, int wv, int nw)
{
  const int row   = lane & 15;
  const int khalf = (lane >> 4) << 3;   // 0 or 8  (A-frag K sub-block)

  for (int mt = wv; mt < mtiles; mt += nw) {
    v8f c = {};
    const _Float16* wr = W + (mt*16 + row)*ld;
    for (int kt = 0; kt < ktiles; ++kt) {
      const int kbase = kt*32;
      v16h a, b;
#pragma unroll
      for (int i=0;i<8;++i){ a[i] = wr[kbase+khalf+i]; a[8+i] = wr[kbase+16+khalf+i]; }
#pragma unroll
      for (int i=0;i<8;++i){ b[i] = x[kbase+khalf+i]; b[8+i] = x[kbase+16+khalf+i]; }
      c = __builtin_amdgcn_wmma_f32_16x16x32_f16(false, a, false, b,
                                                 (short)0, c, false, false);
    }
    if ((lane & 15) == 0) {
      const int base = mt*16 + ((lane >> 4) << 3);
#pragma unroll
      for (int v=0;v<8;++v) y[base+v] = c[v] + bias[base+v];
    }
  }
}

// ---------------------------------------------------------------------------
// Encoder: 4 waves per block, block b in {0,1} handles encoder b.
// All weights live in LDS as f16; 512 sequential DNC steps.
// State out (per encoder, 2048 floats in ws):
//  [0:64] h  [64:128] lr  [128:1152] mem  [1152:1408] link
//  [1408:1424] prec [1424:1440] rw [1440:1456] ww [1456:1472] usage
// ---------------------------------------------------------------------------
__global__ __launch_bounds__(ENC_THREADS) void dnc_encoder_kernel(
    const float* __restrict__ emb0, const float* __restrict__ emb1,
    const float* __restrict__ e1_Wih, const float* __restrict__ e1_Whh,
    const float* __restrict__ e1_bih, const float* __restrict__ e1_bhh,
    const float* __restrict__ e1_iW,  const float* __restrict__ e1_ib,
    const float* __restrict__ e2_Wih, const float* __restrict__ e2_Whh,
    const float* __restrict__ e2_bih, const float* __restrict__ e2_bhh,
    const float* __restrict__ e2_iW,  const float* __restrict__ e2_ib,
    const int* __restrict__ seq1, const int* __restrict__ seq2,
    float* __restrict__ ws)
{
  const int tid  = threadIdx.x;
  const int lane = tid & 31;
  const int wv   = __builtin_amdgcn_readfirstlane(tid >> 5);  // scalar wave id
  const int b = blockIdx.x;
  const float* emb = b ? emb1   : emb0;
  const float* Wih = b ? e2_Wih : e1_Wih;
  const float* Whh = b ? e2_Whh : e1_Whh;
  const float* bih = b ? e2_bih : e1_bih;
  const float* bhh = b ? e2_bhh : e1_bhh;
  const float* iW  = b ? e2_iW  : e1_iW;
  const float* ib  = b ? e2_ib  : e1_ib;
  const int*   seq = b ? seq2   : seq1;
  float* st = ws + b*2048;

  __shared__ _Float16 sWih[192*128];       // 48 KB
  __shared__ _Float16 sWhh[192*64];        // 24 KB
  __shared__ _Float16 sIW[272*64];         // 34 KB (264 rows padded to 272)
  __shared__ float sBih[192], sBhh[192], sIb[272];
  __shared__ float sMem[16*64], sLink[256], sPrec[16], sRw[16], sWw[16], sUsage[16];
  __shared__ float sU[16], sSim[16], sWwN[16], sRwN[16];
  __shared__ _Float16 sX[128], sH16[64], sO16[64];
  __shared__ float sGi[192], sGh[192], sXi[272], sH[64], sLr[64];
  __shared__ float sRk[64], sWk[64], sEv[64], sWv[64];   // precomputed activations

  // one-time: convert weights to f16 in LDS, init state
  for (int k=tid;k<192*128;k+=ENC_THREADS) sWih[k]=(_Float16)Wih[k];
  for (int k=tid;k<192*64; k+=ENC_THREADS) sWhh[k]=(_Float16)Whh[k];
  for (int k=tid;k<272*64; k+=ENC_THREADS) sIW[k]=(_Float16)((k<264*64)?iW[k]:0.0f);
  for (int k=tid;k<192;k+=ENC_THREADS){ sBih[k]=bih[k]; sBhh[k]=bhh[k]; }
  for (int k=tid;k<272;k+=ENC_THREADS) sIb[k]=(k<264)?ib[k]:0.0f;
  for (int k=tid;k<1024;k+=ENC_THREADS) sMem[k]=0.0f;
  for (int k=tid;k<256;k+=ENC_THREADS) sLink[k]=0.0f;
  if (tid<16){ sPrec[tid]=0.0f; sRw[tid]=DELTAV; sWw[tid]=DELTAV; sUsage[tid]=0.0f; }
  for (int k=tid;k<64;k+=ENC_THREADS){ sH[k]=0.0f; sLr[k]=0.0f; }
  __syncthreads();

  for (int t=0;t<L_INV;++t){
    // ---- build GRU input [emb(seq[t]); lr] and h as f16 ----
    const float* er = emb + seq[t]*EMBD;
    for (int i=tid;i<64;i+=ENC_THREADS){
      sX[i]    = (_Float16)er[i];
      sX[64+i] = (_Float16)sLr[i];
      sH16[i]  = (_Float16)sH[i];
    }
    __syncthreads();
    // ---- gi = Wih@x+bih ; gh = Whh@h+bhh (WMMA, tiles split over 4 waves) ----
    wmma_gemv(sWih, 128, sX,   sBih, sGi, 12, 4, lane, wv, ENC_WAVES);
    wmma_gemv(sWhh,  64, sH16, sBhh, sGh, 12, 2, lane, wv, ENC_WAVES);
    __syncthreads();
    // ---- GRU gate fusion (r,z,n) + clip for interface ----
    for (int i=tid;i<64;i+=ENC_THREADS){
      float r = sigf(sGi[i]      + sGh[i]);
      float z = sigf(sGi[64+i]   + sGh[64+i]);
      float n = tanhf(sGi[128+i] + r*sGh[128+i]);
      float h = (1.0f-z)*n + z*sH[i];
      sH[i]  = h;
      sO16[i] = (_Float16)clipf(h);
    }
    __syncthreads();
    // ---- xi = iW@clip(h)+ib (WMMA, rows padded to 272) ----
    wmma_gemv(sIW, 64, sO16, sIb, sXi, 17, 2, lane, wv, ENC_WAVES);
    __syncthreads();

    // ---- interface scalars ----
    const float rs_ = sppf(sXi[64]);
    const float ws_ = sppf(sXi[129]);
    const float fg  = sigf(sXi[258]);
    const float ag  = sigf(sXi[259]);
    const float wg  = sigf(sXi[260]);

    // ---- precompute per-word activations (hoists transcendentals) ----
    for (int i=tid;i<64;i+=ENC_THREADS){
      sRk[i] = tanhf(sXi[i]);        // read key
      sWk[i] = tanhf(sXi[65+i]);     // write key
      sEv[i] = sigf(sXi[130+i]);     // erase vector
      sWv[i] = tanhf(sXi[194+i]);    // write vector
    }
    // ---- usage update (old ww, old rw) ----
    if (tid<16){
      float u = sUsage[tid];
      u = u + (1.0f-u)*sWw[tid];          // R=1 write head
      u = u * (1.0f - fg*sRw[tid]);       // R=1 read head
      sUsage[tid] = u;
      sU[tid] = DELTAV + (1.0f-DELTAV)*u;
    }
    __syncthreads();
    // ---- write content weights: cosine(mem_m, wk)*ws -> softmax ----
    if (tid<16){
      float nb=0.f,na=0.f,dot=0.f;
      for (int wI=0;wI<64;++wI){
        float kv = sWk[wI];
        float mv = sMem[tid*64+wI];
        nb += kv*kv; na += mv*mv; dot += kv*mv;
      }
      sSim[tid] = dot/((sqrtf(na)+DELTAV)*(sqrtf(nb)+DELTAV)) * ws_;
    }
    __syncthreads();
    // ---- allocation (rank-sort of 16 usages) + new write weights ----
    if (tid<16){
      float mx=-1e30f; for(int j=0;j<16;++j) mx=fmaxf(mx,sSim[j]);
      float se=0.f;    for(int j=0;j<16;++j) se+=__expf(sSim[j]-mx);
      float wcw = __expf(sSim[tid]-mx)/se;
      float u = sU[tid];
      float prod = 1.0f;
      for (int j=0;j<16;++j){
        float uj = sU[j];
        bool less = (uj < u) || (uj==u && j<tid);   // stable argsort order
        prod *= less ? uj : 1.0f;
      }
      float alloc = (1.0f-u)*prod;
      sWwN[tid] = wg*(ag*alloc + (1.0f-ag)*wcw);
    }
    __syncthreads();
    // ---- memory erase/write (new ww) ----
    for (int k=tid;k<1024;k+=ENC_THREADS){
      int m = k>>6, wI = k&63;
      float wwm = sWwN[m];
      sMem[k] = sMem[k]*(1.0f - wwm*sEv[wI]) + wwm*sWv[wI];
    }
    // ---- temporal link (new ww, old prec) ----
    for (int k=tid;k<256;k+=ENC_THREADS){
      int i=k>>4, j=k&15;
      float wi=sWwN[i], wj=sWwN[j];
      float nl = (1.0f-wi-wj)*sLink[k] + wi*sPrec[j];
      sLink[k] = (i==j)?0.0f:nl;
    }
    __syncthreads();
    // ---- precedence ----
    if (tid<16){
      float s=0.f; for(int j=0;j<16;++j) s+=sWwN[j];
      sPrec[tid] = (1.0f-s)*sPrec[tid] + sWwN[tid];
    }
    __syncthreads();
    // ---- read: content weights with rk/rs (post-write memory) ----
    if (tid<16){
      float nb=0.f,na=0.f,dot=0.f;
      for (int wI=0;wI<64;++wI){
        float kv = sRk[wI];
        float mv = sMem[tid*64+wI];
        nb += kv*kv; na += mv*mv; dot += kv*mv;
      }
      sSim[tid] = dot/((sqrtf(na)+DELTAV)*(sqrtf(nb)+DELTAV)) * rs_;
    }
    __syncthreads();
    // ---- forward/backward weightings + mode mix (old rw) ----
    if (tid<16){
      float mx=-1e30f; for(int j=0;j<16;++j) mx=fmaxf(mx,sSim[j]);
      float se=0.f;    for(int j=0;j<16;++j) se+=__expf(sSim[j]-mx);
      float cw = __expf(sSim[tid]-mx)/se;
      float fw=0.f, bw=0.f;
      for (int j=0;j<16;++j){ fw += sLink[tid*16+j]*sRw[j]; bw += sLink[j*16+tid]*sRw[j]; }
      float a0=sXi[261],a1=sXi[262],a2=sXi[263];
      float m3=fmaxf(a0,fmaxf(a1,a2));
      float e0=__expf(a0-m3),e1=__expf(a1-m3),e2=__expf(a2-m3);
      float s3=e0+e1+e2;
      sRwN[tid] = (e0/s3)*bw + (e1/s3)*fw + (e2/s3)*cw;
    }
    __syncthreads();
    if (tid<16){ sRw[tid]=sRwN[tid]; sWw[tid]=sWwN[tid]; }
    __syncthreads();
    // ---- read vector ----
    for (int wI=tid;wI<64;wI+=ENC_THREADS){
      float s=0.f;
      for (int m=0;m<16;++m) s += sRw[m]*sMem[m*64+wI];
      sLr[wI]=s;
    }
    __syncthreads();
  }

  // ---- hand off state to decoder through workspace ----
  for (int i=tid;i<64;i+=ENC_THREADS){ st[i]=sH[i]; st[64+i]=sLr[i]; }
  for (int k=tid;k<1024;k+=ENC_THREADS) st[128+k]=sMem[k];
  for (int k=tid;k<256;k+=ENC_THREADS)  st[1152+k]=sLink[k];
  if (tid<16){ st[1408+tid]=sPrec[tid]; st[1424+tid]=sRw[tid];
               st[1440+tid]=sWw[tid];   st[1456+tid]=sUsage[tid]; }
}

// ---------------------------------------------------------------------------
// decoder memory read (M=16 cells, W=64)
// ---------------------------------------------------------------------------
__device__ __forceinline__ void dec_read(const float* __restrict__ mem,
                                         const float* __restrict__ link,
                                         float* __restrict__ rw,
                                         const float* __restrict__ xiK,
                                         float strengthRaw,
                                         const float* __restrict__ modes3,
                                         float* __restrict__ rout,
                                         float* __restrict__ sSim,
                                         float* __restrict__ sRwN, int tid)
{
  const float st = sppf(strengthRaw);
  if (tid<16){
    float nb=0.f,na=0.f,dot=0.f;
    for (int wI=0;wI<64;++wI){
      float kv = tanhf(xiK[wI]);
      float mv = mem[tid*64+wI];
      nb += kv*kv; na += mv*mv; dot += kv*mv;
    }
    sSim[tid] = dot/((sqrtf(na)+DELTAV)*(sqrtf(nb)+DELTAV)) * st;
  }
  __syncthreads();
  if (tid<16){
    float mx=-1e30f; for(int j=0;j<16;++j) mx=fmaxf(mx,sSim[j]);
    float se=0.f;    for(int j=0;j<16;++j) se+=__expf(sSim[j]-mx);
    float cw = __expf(sSim[tid]-mx)/se;
    float fw=0.f,bw=0.f;
    for (int j=0;j<16;++j){ fw += link[tid*16+j]*rw[j]; bw += link[j*16+tid]*rw[j]; }
    float a0=modes3[0],a1=modes3[1],a2=modes3[2];
    float m3=fmaxf(a0,fmaxf(a1,a2));
    float e0=__expf(a0-m3),e1=__expf(a1-m3),e2=__expf(a2-m3);
    float s3=e0+e1+e2;
    sRwN[tid] = (e0/s3)*bw + (e1/s3)*fw + (e2/s3)*cw;
  }
  __syncthreads();
  if (tid<16) rw[tid]=sRwN[tid];
  __syncthreads();
  for (int wI=tid;wI<64;wI+=ENC_THREADS){
    float s=0.f;
    for (int m=0;m<16;++m) s += rw[m]*mem[m*64+wI];
    rout[wI]=s;
  }
  __syncthreads();
}

// ---------------------------------------------------------------------------
// Decoder: 4 waves, 65 steps, f32 weights streamed from L2 (prefetched).
// ---------------------------------------------------------------------------
__global__ __launch_bounds__(ENC_THREADS) void dnc_decoder_kernel(
    const float* __restrict__ emb2,
    const float* __restrict__ dWih, const float* __restrict__ dWhh,
    const float* __restrict__ dbih, const float* __restrict__ dbhh,
    const float* __restrict__ ifwW, const float* __restrict__ ifwB,
    const float* __restrict__ r2oW, const float* __restrict__ r2oB,
    const float* __restrict__ outW, const float* __restrict__ outB,
    const int* __restrict__ seq3,
    const float* __restrict__ st1, const float* __restrict__ st2,
    float* __restrict__ out)
{
  const int tid = threadIdx.x;
  __shared__ float sM1[1024], sM2[1024], sL1[256], sL2[256], sRw1[16], sRw2[16];
  __shared__ float sH[128], sR1[64], sR2[64], sX[192];
  __shared__ float sGi[384], sGh[384], sXi[136], sOH[128];
  __shared__ float sSim[16], sRwN[16];

  for (int k=tid;k<1024;k+=ENC_THREADS){ sM1[k]=st1[128+k]; sM2[k]=st2[128+k]; }
  for (int k=tid;k<256;k+=ENC_THREADS){ sL1[k]=st1[1152+k]; sL2[k]=st2[1152+k]; }
  if (tid<16){ sRw1[tid]=st1[1424+tid]; sRw2[tid]=st2[1424+tid]; }
  for (int k=tid;k<64;k+=ENC_THREADS){
    sH[k]=st1[k]; sH[64+k]=st2[k]; sR1[k]=st1[64+k]; sR2[k]=st2[64+k];
  }
  // pull decoder weights into cache (global_prefetch_b8)
  for (int off=tid*64; off<384*192; off+=ENC_THREADS*64) __builtin_prefetch(dWih+off,0,0);
  for (int off=tid*64; off<384*128; off+=ENC_THREADS*64) __builtin_prefetch(dWhh+off,0,0);
  for (int off=tid*64; off<138*128; off+=ENC_THREADS*64) __builtin_prefetch(outW+off,0,0);
  __syncthreads();

  for (int t=0;t<=L_OUTV;++t){
    const int tok = (t==0)? TOKEN_STARTV : seq3[t-1];
    const float* te = emb2 + tok*EMBD;
    for (int i=tid;i<64;i+=ENC_THREADS){ sX[i]=te[i]; sX[64+i]=sR1[i]; sX[128+i]=sR2[i]; }
    __syncthreads();
    // GRU gates: gi(384x192), gh(384x128)
    for (int rI=tid; rI<384; rI+=ENC_THREADS){
      const float* w = dWih + rI*192;
      float s = dbih[rI];
      for (int k=0;k<192;++k) s += w[k]*sX[k];
      sGi[rI]=s;
      const float* w2 = dWhh + rI*128;
      float s2 = dbhh[rI];
      for (int k=0;k<128;++k) s2 += w2[k]*sH[k];
      sGh[rI]=s2;
    }
    __syncthreads();
    for (int i=tid;i<128;i+=ENC_THREADS){
      float r=sigf(sGi[i]      + sGh[i]);
      float z=sigf(sGi[128+i]  + sGh[128+i]);
      float n=tanhf(sGi[256+i] + r*sGh[256+i]);
      sH[i]=(1.0f-z)*n + z*sH[i];
    }
    __syncthreads();
    // interface xi(136x128)
    for (int rI=tid;rI<136;rI+=ENC_THREADS){
      const float* w=ifwW+rI*128;
      float s=ifwB[rI];
      for (int k=0;k<128;++k) s+=w[k]*sH[k];
      sXi[rI]=s;
    }
    __syncthreads();
    // two memory reads
    dec_read(sM1,sL1,sRw1,&sXi[0],  sXi[128], &sXi[130], sR1, sSim, sRwN, tid);
    dec_read(sM2,sL2,sRw2,&sXi[64], sXi[129], &sXi[133], sR2, sSim, sRwN, tid);
    // o = r2o@[r1;r2]+b, then (o+h)
    for (int rI=tid;rI<128;rI+=ENC_THREADS){
      const float* w=r2oW+rI*128;
      float s=r2oB[rI];
      for (int k=0;k<64;++k) s += w[k]*sR1[k] + w[64+k]*sR2[k];
      sOH[rI]=s+sH[rI];
    }
    __syncthreads();
    // logits(138x128)
    for (int rI=tid;rI<138;rI+=ENC_THREADS){
      const float* w=outW+rI*128;
      float s=outB[rI];
      for (int k=0;k<128;++k) s += w[k]*sOH[k];
      out[t*138+rI]=s;
    }
    __syncthreads();
  }
}

extern "C" void kernel_launch(void* const* d_in, const int* in_sizes, int n_in,
                              void* d_out, int out_size, void* d_ws, size_t ws_size,
                              hipStream_t stream) {
  (void)in_sizes; (void)n_in; (void)out_size; (void)ws_size;
  const float* emb0    = (const float*)d_in[0];
  const float* emb1    = (const float*)d_in[1];
  const float* emb2    = (const float*)d_in[2];
  const float* e1_Wih  = (const float*)d_in[3];
  const float* e1_Whh  = (const float*)d_in[4];
  const float* e1_bih  = (const float*)d_in[5];
  const float* e1_bhh  = (const float*)d_in[6];
  const float* e1_iW   = (const float*)d_in[7];
  const float* e1_ib   = (const float*)d_in[8];
  const float* e2_Wih  = (const float*)d_in[9];
  const float* e2_Whh  = (const float*)d_in[10];
  const float* e2_bih  = (const float*)d_in[11];
  const float* e2_bhh  = (const float*)d_in[12];
  const float* e2_iW   = (const float*)d_in[13];
  const float* e2_ib   = (const float*)d_in[14];
  const float* dec_Wih = (const float*)d_in[15];
  const float* dec_Whh = (const float*)d_in[16];
  const float* dec_bih = (const float*)d_in[17];
  const float* dec_bhh = (const float*)d_in[18];
  const float* ifw_W   = (const float*)d_in[19];
  const float* ifw_b   = (const float*)d_in[20];
  const float* r2o_W   = (const float*)d_in[21];
  const float* r2o_b   = (const float*)d_in[22];
  const float* out_W   = (const float*)d_in[23];
  const float* out_b   = (const float*)d_in[24];
  const int*   seq1    = (const int*)d_in[25];
  const int*   seq2    = (const int*)d_in[26];
  const int*   seq3    = (const int*)d_in[27];

  float* ws = (float*)d_ws;

  dnc_encoder_kernel<<<2, ENC_THREADS, 0, stream>>>(
      emb0, emb1,
      e1_Wih, e1_Whh, e1_bih, e1_bhh, e1_iW, e1_ib,
      e2_Wih, e2_Whh, e2_bih, e2_bhh, e2_iW, e2_ib,
      seq1, seq2, ws);

  dnc_decoder_kernel<<<1, ENC_THREADS, 0, stream>>>(
      emb2, dec_Wih, dec_Whh, dec_bih, dec_bhh,
      ifw_W, ifw_b, r2o_W, r2o_b, out_W, out_b,
      seq3, ws, ws + 2048, (float*)d_out);
}
